// FuzzyNet_45140106280988
// MI455X (gfx1250) — matrix operand order repaired and checked by reading, not compiled, not run on hardware
//
#include <hip/hip_runtime.h>
#include <hip/hip_bf16.h>
#include <cstdint>

// ---------------------------------------------------------------------------
// Types for WMMA fragments (gfx1250, wave32)
// ---------------------------------------------------------------------------
typedef __attribute__((ext_vector_type(16))) __bf16        v16bf;
typedef __attribute__((ext_vector_type(16))) unsigned short v16u;
typedef __attribute__((ext_vector_type(8)))  float          v8f;

#define BN_EPS 1e-5f

// ---------------------------------------------------------------------------
// Implicit-GEMM convolution with WMMA bf16 (fp32 accumulate).
//   M = Cout, N = B*H*W, K = Cin*k*k, SAME padding (pad = k/2).
//   Block = 128 threads = 4 waves. Block tile: 64(M) x 64(N), K-stage 64.
//   A (weights) panel: CDNA5 async global->LDS (b128 fast path, ASYNCcnt).
//   B (im2col) panel: div-free incremental decode, ds stores.
//   Each wave: 4 accumulators, 8 v_wmma per LDS stage.
// ---------------------------------------------------------------------------
#define CONV_BM 64
#define CONV_BN 64
#define CONV_BK 64
#define LPAD 4   // row pad (floats): pitch 68*4B = 272B, 16B aligned, bank-skewed

__global__ __launch_bounds__(128)
void conv_wmma_kernel(const float* __restrict__ x, const float* __restrict__ w,
                      const float* __restrict__ bias, const float* __restrict__ res,
                      float* __restrict__ out,
                      int B, int Cin, int H, int W, int Cout, int ksz, int pad,
                      int relu_flag)
{
    const int HW   = H * W;
    const int Npix = B * HW;
    const int K    = Cin * ksz * ksz;
    const int n0   = blockIdx.x * CONV_BN;
    const int m0   = blockIdx.y * CONV_BM;
    const int tid  = threadIdx.x;
    const int lane = tid & 31;
    const int wave = tid >> 5;

    __shared__ __align__(16) float Asf[CONV_BM][CONV_BK + LPAD]; // [m][k]
    __shared__ __align__(16) float Bsf[CONV_BK][CONV_BN + LPAD]; // [k][n]

    v8f acc[4] = {};

    const int ml   = lane & 15;
    const int half = lane >> 4;
    const int arow = (wave << 4) + ml;

    // ---- per-thread fixed panel assignments ----
    const int a_m    = tid & 63;            // A row (m in block)
    const int a_k0   = (tid >> 6) << 5;     // A col base (0 / 32)
    const int a_oc   = m0 + a_m;
    const bool a_ok  = (a_oc < Cout);
    const int b_k    = tid & 63;            // B row (k in stage)
    const int b_nb   = (tid >> 6) << 5;     // B col base (0 / 32)
    const bool kvec  = ((K & 3) == 0);      // 16B-aligned async fast path

    // decode pixel for b_nb once; advanced incrementally inside the loop
    const int ng0 = n0 + b_nb;
    int pb = ng0 / HW;
    int pp = ng0 - pb * HW;
    int pyy = pp / W;
    int pxx = pp - pyy * W;
    const int kk2 = ksz * ksz;

    for (int kk = 0; kk < K; kk += CONV_BK) {
        // ---- A panel: async global -> LDS (no VGPR round-trip) ----
        {
            unsigned ldsrow = (unsigned)(size_t)&Asf[a_m][a_k0];
            const float* gp = w + (long)a_oc * K + kk + a_k0;
            if (a_ok && kvec && (kk + a_k0 + 32 <= K)) {
#pragma unroll
                for (int t = 0; t < 8; ++t) {
                    unsigned lo = ldsrow + t * 16;
                    unsigned long long ga = (unsigned long long)(size_t)(gp + t * 4);
                    asm volatile("global_load_async_to_lds_b128 %0, %1, off"
                                 :: "v"(lo), "v"(ga) : "memory");
                }
            } else {
#pragma unroll 4
                for (int t = 0; t < 32; ++t) {
                    int kg = kk + a_k0 + t;
                    if (a_ok && kg < K) {
                        unsigned lo = ldsrow + t * 4;
                        unsigned long long ga = (unsigned long long)(size_t)(gp + t);
                        asm volatile("global_load_async_to_lds_b32 %0, %1, off"
                                     :: "v"(lo), "v"(ga) : "memory");
                    } else {
                        Asf[a_m][a_k0 + t] = 0.f;
                    }
                }
            }
        }
        // ---- B panel: im2col row, division-free inner loop ----
        {
            int kg = kk + b_k;
            bool kok = (kg < K);
            int ic = 0, r = 0, s = 0;
            if (kok) {
                ic = kg / kk2;                 // once per stage
                int rem = kg - ic * kk2;
                r = (ksz == 3) ? ((rem * 11) >> 5) : 0;   // rem/3 for rem<9
                s = rem - r * ksz;
            }
            int b = pb, yy = pyy, xx = pxx, ng = ng0;
            const long icb = ((long)ic) * H;
#pragma unroll 4
            for (int j = 0; j < 32; ++j) {
                float v = 0.f;
                if (kok && ng < Npix) {
                    int iy = yy + r - pad, ix = xx + s - pad;
                    if (iy >= 0 && iy < H && ix >= 0 && ix < W)
                        v = x[(((long)b * Cin) * H + icb + iy) * W + ix];
                }
                Bsf[b_k][b_nb + j] = v;
                ++ng; ++xx;
                if (xx == W) { xx = 0; ++yy; if (yy == H) { yy = 0; ++b; } }
            }
        }
        asm volatile("s_wait_asynccnt 0x0" ::: "memory");
        __syncthreads();

        // ---- two 32-wide K chunks per stage; 4 N tiles per wave ----
#pragma unroll
        for (int kc = 0; kc < 2; ++kc) {
            const int kb = kc << 5;
            v16u au;
#pragma unroll
            for (int e = 0; e < 16; ++e) {
                // ISA 16-bit A layout: (e<8?0:16) + half*8 + (e&7)
                int ka = kb + ((e & 8) << 1) + (half << 3) + (e & 7);
                au[e] = (unsigned short)(__builtin_bit_cast(unsigned int, Asf[arow][ka]) >> 16);
            }
            v16bf a = __builtin_bit_cast(v16bf, au);
#pragma unroll
            for (int nt = 0; nt < 4; ++nt) {
                v16u bu;
#pragma unroll
                for (int e = 0; e < 16; ++e)
                    bu[e] = (unsigned short)(__builtin_bit_cast(
                                unsigned int, Bsf[kb + ((e & 8) << 1) + (half << 3) + (e & 7)][(nt << 4) + ml]) >> 16);
                v16bf b = __builtin_bit_cast(v16bf, bu);
                acc[nt] = __builtin_amdgcn_wmma_f32_16x16x32_bf16(
                              false, a, false, b, (short)0, acc[nt], false, false);
            }
        }
        __syncthreads();
    }

    // ---- epilogue: C layout VGPR j -> M=j (lanes 0-15) / M=j+8 (lanes 16-31)
    const int mo = half << 3;
#pragma unroll
    for (int nt = 0; nt < 4; ++nt) {
        int n = n0 + (nt << 4) + ml;
        if (n < Npix) {
            int b = n / HW;
            int p = n - b * HW;
            int yy = p / W, xx = p - yy * W;
#pragma unroll
            for (int j = 0; j < 8; ++j) {
                int m = m0 + (wave << 4) + j + mo;
                if (m < Cout) {
                    float v = acc[nt][j] + (bias ? bias[m] : 0.f);
                    long oi = (((long)b * Cout + m) * H + yy) * W + xx;
                    if (res) v += res[oi];
                    if (relu_flag) v = fmaxf(v, 0.f);
                    out[oi] = v;
                }
            }
        }
    }
}

// ---------------------------------------------------------------------------
// BatchNorm (training-mode): pass 1 per-channel mean / biased variance.
// ---------------------------------------------------------------------------
__global__ __launch_bounds__(256)
void bn_reduce_kernel(const float* __restrict__ x, float* __restrict__ mean,
                      float* __restrict__ var, int B, int C, int HW)
{
    int c = blockIdx.x;
    long per = (long)B * HW;
    float s = 0.f, s2 = 0.f;
    for (long i = threadIdx.x; i < per; i += 256) {
        long b = i / HW, p = i - b * HW;
        float v = x[((long)b * C + c) * HW + p];
        s += v; s2 += v * v;
    }
    __shared__ float rs[256], rq[256];
    rs[threadIdx.x] = s; rq[threadIdx.x] = s2;
    __syncthreads();
    for (int o = 128; o > 0; o >>= 1) {
        if (threadIdx.x < o) { rs[threadIdx.x] += rs[threadIdx.x + o];
                               rq[threadIdx.x] += rq[threadIdx.x + o]; }
        __syncthreads();
    }
    if (threadIdx.x == 0) {
        float m = rs[0] / (float)per;
        mean[c] = m;
        var[c]  = fmaxf(rq[0] / (float)per - m * m, 0.f);
    }
}

// pass 2: normalize + scale/shift, optional fused residual-add and ReLU.
__global__ void bn_apply_kernel(const float* __restrict__ x,
                                const float* __restrict__ mean,
                                const float* __restrict__ var,
                                const float* __restrict__ g,
                                const float* __restrict__ bt,
                                const float* __restrict__ res,
                                float* __restrict__ out,
                                int C, int HW, long total, int relu_flag)
{
    long i = (long)blockIdx.x * blockDim.x + threadIdx.x;
    if (i >= total) return;
    int c = (int)((i / HW) % C);
    float v = (x[i] - mean[c]) * rsqrtf(var[c] + BN_EPS) * g[c] + bt[c];
    if (res) v += res[i];
    if (relu_flag) v = fmaxf(v, 0.f);
    out[i] = v;
}

// ---------------------------------------------------------------------------
// maxpool2 with optional fused elementwise add (maxpool2(h + g)).
// ---------------------------------------------------------------------------
__global__ void maxpool_add_kernel(const float* __restrict__ x,
                                   const float* __restrict__ g,
                                   float* __restrict__ out,
                                   int B, int C, int H, int W)
{
    int Ho = H >> 1, Wo = W >> 1;
    long total = (long)B * C * Ho * Wo;
    long i = (long)blockIdx.x * blockDim.x + threadIdx.x;
    if (i >= total) return;
    long t = i;
    int xo = (int)(t % Wo); t /= Wo;
    int yo = (int)(t % Ho); t /= Ho;          // t = b*C + c
    long base = t * (long)H * W;
    float m = -3.4e38f;
#pragma unroll
    for (int dy = 0; dy < 2; ++dy)
#pragma unroll
        for (int dx = 0; dx < 2; ++dx) {
            long idx = base + (long)(2 * yo + dy) * W + (2 * xo + dx);
            float v = x[idx] + (g ? g[idx] : 0.f);
            m = fmaxf(m, v);
        }
    out[i] = m;
}

// ---------------------------------------------------------------------------
// ConvTranspose2d(kernel=2, stride=2, bias=False), optional fused skip add.
//   w: [I, O, 2, 2]
// ---------------------------------------------------------------------------
__global__ void deconv2x2_kernel(const float* __restrict__ x,
                                 const float* __restrict__ wt,
                                 const float* __restrict__ res,
                                 float* __restrict__ out,
                                 int B, int I, int O, int H, int W)
{
    int Ho = 2 * H, Wo = 2 * W;
    long total = (long)B * O * Ho * Wo;
    long idx = (long)blockIdx.x * blockDim.x + threadIdx.x;
    if (idx >= total) return;
    long t = idx;
    int X = (int)(t % Wo); t /= Wo;
    int Y = (int)(t % Ho); t /= Ho;
    int o = (int)(t % O);
    int b = (int)(t / O);
    int h = Y >> 1, k = Y & 1, w2 = X >> 1, l = X & 1;
    long HWs = (long)H * W;
    const float* xp = x + ((long)b * I * H + h) * W + w2;
    const float* wp = wt + ((long)o * 2 + k) * 2 + l;
    float s = 0.f;
    for (int i = 0; i < I; ++i)
        s += xp[(long)i * HWs] * wp[(long)i * O * 4];
    if (res) s += res[idx];
    out[idx] = s;
}

// ---------------------------------------------------------------------------
// fuzzy membership: n==0 -> exp(0)=1; n==1 -> exp(-((x1-mu0)/sigma0)^2)
// ---------------------------------------------------------------------------
__global__ void fuzzy_m_kernel(const float* __restrict__ x1,
                               const float* __restrict__ mu,
                               const float* __restrict__ sigma,
                               int nfuzzy, float* __restrict__ out, long total)
{
    long i = (long)blockIdx.x * blockDim.x + threadIdx.x;
    if (i >= total) return;
    if (nfuzzy == 0) { out[i] = 1.f; return; }
    float d = (x1[i] - mu[0]) / sigma[0];
    out[i] = __expf(-d * d);
}

// elementwise add
__global__ void add_kernel(const float* __restrict__ a, const float* __restrict__ b,
                           float* __restrict__ out, long total)
{
    long i = (long)blockIdx.x * blockDim.x + threadIdx.x;
    if (i >= total) return;
    out[i] = a[i] + b[i];
}

// ---------------------------------------------------------------------------
// Host-side launch helpers (all on `stream`; graph-capture safe)
// ---------------------------------------------------------------------------
static void L_conv(const float* x, const float* w, const float* b, const float* res,
                   float* out, int B, int Cin, int H, int W, int Cout, int k, int pad,
                   int relu, hipStream_t s)
{
    int Npix = B * H * W;
    dim3 grid((Npix + CONV_BN - 1) / CONV_BN, (Cout + CONV_BM - 1) / CONV_BM);
    conv_wmma_kernel<<<grid, 128, 0, s>>>(x, w, b, res, out, B, Cin, H, W, Cout, k, pad, relu);
}

static void L_bn(const float* x, const float* g, const float* b, const float* res,
                 float* out, int B, int C, int H, int W, int relu, float* stats,
                 hipStream_t s)
{
    bn_reduce_kernel<<<C, 256, 0, s>>>(x, stats, stats + C, B, C, H * W);
    long total = (long)B * C * H * W;
    unsigned blocks = (unsigned)((total + 255) / 256);
    bn_apply_kernel<<<blocks, 256, 0, s>>>(x, stats, stats + C, g, b, res, out,
                                           C, H * W, total, relu);
}

static void L_pool(const float* x, const float* g, float* out, int B, int C,
                   int H, int W, hipStream_t s)
{
    long total = (long)B * C * (H / 2) * (W / 2);
    maxpool_add_kernel<<<(unsigned)((total + 255) / 256), 256, 0, s>>>(x, g, out, B, C, H, W);
}

static void L_deconv(const float* x, const float* w, const float* res, float* out,
                     int B, int I, int O, int H, int W, hipStream_t s)
{
    long total = (long)B * O * (2 * H) * (2 * W);
    deconv2x2_kernel<<<(unsigned)((total + 255) / 256), 256, 0, s>>>(x, w, res, out, B, I, O, H, W);
}

// fuzzy(x, fp) -> out  (out shape = [B,C,H,W]); fzA/fzB are [B*H*W] scratch
static void L_fuzzy(const float* x, int B, int C, int H, int W, int nfuzzy,
                    const float* c1w, const float* c1b, const float* c2w, const float* c2b,
                    const float* mu, const float* sigma,
                    const float* bn1g, const float* bn1b,
                    const float* bn2g, const float* bn2b,
                    float* out, float* fzA, float* fzB, float* stats, hipStream_t s)
{
    long px = (long)B * H * W;
    L_conv(x, c1w, c1b, nullptr, fzA, B, C, H, W, 1, 3, 1, 0, s);           // [B,1,H,W]
    fuzzy_m_kernel<<<(unsigned)((px + 255) / 256), 256, 0, s>>>(fzA, mu, sigma, nfuzzy, fzB, px);
    L_bn(fzB, bn1g, bn1b, nullptr, fzA, B, 1, H, W, 0, stats, s);           // bn1
    L_conv(fzA, c2w, c2b, nullptr, out, B, 1, H, W, C, 3, 1, 0, s);         // [B,C,H,W]
    L_bn(out, bn2g, bn2b, nullptr, out, B, C, H, W, 0, stats, s);           // bn2 (in place)
}

// ---------------------------------------------------------------------------
extern "C" void kernel_launch(void* const* d_in, const int* in_sizes, int n_in,
                              void* d_out, int out_size, void* d_ws, size_t ws_size,
                              hipStream_t stream)
{
    (void)in_sizes; (void)n_in; (void)out_size; (void)ws_size;
    const int B = 4;
    #define P(i) ((const float*)d_in[(i)])
    const float* X = P(0);

    // ---- workspace arena (floats) ----
    float* ws = (float*)d_ws;
    size_t off = 0;
    auto alloc = [&](size_t n) { float* p = ws + off; off += n; return p; };
    float* t224a = alloc(12845056); // 64@224
    float* t224b = alloc(12845056);
    float* c1    = alloc(3211264);  // 64@112
    float* f1    = alloc(3211264);
    float* t112a = alloc(6422528);  // 128@112 (also holds 64@112 fuzzy temp)
    float* t112b = alloc(6422528);
    float* gA112 = alloc(6422528);
    float* gB112 = alloc(6422528);
    float* c2    = alloc(1605632);  // 128@56
    float* f2    = alloc(1605632);
    float* t56a  = alloc(3211264);  // 256@56
    float* t56b  = alloc(3211264);
    float* gA56  = alloc(3211264);
    float* gB56  = alloc(3211264);
    float* c3    = alloc(802816);   // 256@28
    float* f3    = alloc(802816);
    float* t28a  = alloc(1605632);  // 512@28
    float* t28b  = alloc(1605632);
    float* gA28  = alloc(1605632);
    float* gB28  = alloc(1605632);
    float* h4    = alloc(401408);   // 512@14
    float* f4    = alloc(401408);
    float* t14a  = alloc(401408);
    float* t14b  = alloc(401408);
    float* fzA   = alloc(50176);    // [B*112*112] max
    float* fzB   = alloc(50176);
    float* stats = alloc(1024);     // bn mean|var (C<=512)

    // ================= encoder stage 1 =================
    L_conv(X, P(1), P(2), nullptr, t224a, B, 3, 224, 224, 64, 3, 1, 1, stream);     // conv1_1+relu
    L_conv(t224a, P(3), P(4), nullptr, t224b, B, 64, 224, 224, 64, 3, 1, 1, stream);// conv1_2+relu
    L_pool(t224b, nullptr, c1, B, 64, 224, 224, stream);                            // c1: 64@112

    // f1 = bn(fuzzy(c1, fuzzy_1), fbn1) + c1     (fuzzy_1 is at params idx 71..)
    L_fuzzy(c1, B, 64, 112, 112, 0, P(71), P(72), P(73), P(74), P(75), P(76),
            P(77), P(78), P(79), P(80), t112a, fzA, fzB, stats, stream);
    L_bn(t112a, P(55), P(56), c1, f1, B, 64, 112, 112, 0, stats, stream);

    // ---- stage 2 residual (g) path ----
    L_conv(c1, P(9),  P(10), nullptr, gA112, B, 64, 112, 112, 64, 1, 0, 0, stream); // conv2_r1
    L_bn(gA112, P(15), P(16), nullptr, gA112, B, 64, 112, 112, 1, stats, stream);   // bn2_r1+relu
    L_conv(gA112, P(11), P(12), nullptr, gB112, B, 64, 112, 112, 128, 3, 1, 0, stream); // conv2_r2
    L_bn(gB112, P(17), P(18), nullptr, gB112, B, 128, 112, 112, 1, stats, stream);  // bn2_r2+relu
    L_conv(gB112, P(13), P(14), nullptr, gA112, B, 128, 112, 112, 128, 1, 0, 0, stream); // conv2_r3
    // ---- stage 2 main path ----
    L_conv(c1, P(5), P(6), nullptr, t112a, B, 64, 112, 112, 128, 3, 1, 1, stream);  // conv2_1+relu
    L_conv(t112a, P(7), P(8), nullptr, t112b, B, 128, 112, 112, 128, 3, 1, 1, stream); // conv2_2+relu
    L_pool(t112b, gA112, c2, B, 128, 112, 112, stream);                             // c2: 128@56

    // f2 = bn(fuzzy(c2, fuzzy_2), fbn2) + c2
    L_fuzzy(c2, B, 128, 56, 56, 0, P(81), P(82), P(83), P(84), P(85), P(86),
            P(87), P(88), P(89), P(90), t56a, fzA, fzB, stats, stream);
    L_bn(t56a, P(57), P(58), c2, f2, B, 128, 56, 56, 0, stats, stream);

    // ---- stage 3 residual path ----
    L_conv(c2, P(25), P(26), nullptr, gA56, B, 128, 56, 56, 128, 1, 0, 0, stream);  // conv3_r1
    L_bn(gA56, P(31), P(32), nullptr, gA56, B, 128, 56, 56, 1, stats, stream);      // bn3_r1+relu
    L_conv(gA56, P(27), P(28), nullptr, gB56, B, 128, 56, 56, 256, 3, 1, 0, stream);// conv3_r2
    L_bn(gB56, P(33), P(34), nullptr, gB56, B, 256, 56, 56, 1, stats, stream);      // bn3_r2+relu
    L_conv(gB56, P(29), P(30), nullptr, gA56, B, 256, 56, 56, 256, 1, 0, 0, stream);// conv3_r3
    // ---- stage 3 main path ----
    L_conv(c2, P(19), P(20), nullptr, t56a, B, 128, 56, 56, 256, 3, 1, 1, stream);  // conv3_1+relu
    L_conv(t56a, P(21), P(22), nullptr, t56b, B, 256, 56, 56, 256, 3, 1, 1, stream);// conv3_2+relu
    L_conv(t56b, P(23), P(24), nullptr, t56a, B, 256, 56, 56, 256, 3, 1, 1, stream);// conv3_3+relu
    L_pool(t56a, gA56, c3, B, 256, 56, 56, stream);                                 // c3: 256@28

    // f3 = bn(fuzzy(c3, fuzzy_3), fbn3) + c3
    L_fuzzy(c3, B, 256, 28, 28, 0, P(91), P(92), P(93), P(94), P(95), P(96),
            P(97), P(98), P(99), P(100), t28a, fzA, fzB, stats, stream);
    L_bn(t28a, P(59), P(60), c3, f3, B, 256, 28, 28, 0, stats, stream);

    // ---- stage 4 residual path ----
    L_conv(c3, P(41), P(42), nullptr, gA28, B, 256, 28, 28, 256, 1, 0, 0, stream);  // conv4_r1
    L_bn(gA28, P(47), P(48), nullptr, gA28, B, 256, 28, 28, 1, stats, stream);      // bn4_r1+relu
    L_conv(gA28, P(43), P(44), nullptr, gB28, B, 256, 28, 28, 512, 3, 1, 0, stream);// conv4_r2
    L_bn(gB28, P(49), P(50), nullptr, gB28, B, 512, 28, 28, 1, stats, stream);      // bn4_r2+relu
    L_conv(gB28, P(45), P(46), nullptr, gA28, B, 512, 28, 28, 512, 1, 0, 0, stream);// conv4_r3
    // ---- stage 4 main path ----
    L_conv(c3, P(35), P(36), nullptr, t28a, B, 256, 28, 28, 512, 3, 1, 1, stream);  // conv4_1+relu
    L_conv(t28a, P(37), P(38), nullptr, t28b, B, 512, 28, 28, 512, 3, 1, 1, stream);// conv4_2+relu
    L_conv(t28b, P(39), P(40), nullptr, t28a, B, 512, 28, 28, 512, 3, 1, 1, stream);// conv4_3+relu
    L_pool(t28a, gA28, h4, B, 512, 28, 28, stream);                                 // h4: 512@14

    // f4 = bn(fuzzy(h4, fuzzy_4 [n=1]), fbn4)
    L_fuzzy(h4, B, 512, 14, 14, 1, P(101), P(102), P(103), P(104), P(105), P(106),
            P(107), P(108), P(109), P(110), t14a, fzA, fzB, stats, stream);
    L_bn(t14a, P(61), P(62), nullptr, f4, B, 512, 14, 14, 0, stats, stream);

    // h = bn(f4 + h4, bn1)
    {
        long tot = (long)B * 512 * 14 * 14;
        add_kernel<<<(unsigned)((tot + 255) / 256), 256, 0, stream>>>(f4, h4, t14b, tot);
    }
    L_bn(t14b, P(63), P(64), nullptr, t14b, B, 512, 14, 14, 0, stats, stream);

    // ================= decoder =================
    L_deconv(t14b, P(51), f3, gA28, B, 512, 256, 14, 14, stream);   // +f3 -> 256@28
    L_bn(gA28, P(65), P(66), nullptr, gA28, B, 256, 28, 28, 0, stats, stream);
    L_deconv(gA28, P(52), f2, gA56, B, 256, 128, 28, 28, stream);   // +f2 -> 128@56
    L_bn(gA56, P(67), P(68), nullptr, gA56, B, 128, 56, 56, 0, stats, stream);
    L_deconv(gA56, P(53), f1, gA112, B, 128, 64, 56, 56, stream);   // +f1 -> 64@112
    L_bn(gA112, P(69), P(70), nullptr, gA112, B, 64, 112, 112, 0, stats, stream);
    L_deconv(gA112, P(54), nullptr, (float*)d_out, B, 64, 2, 112, 112, stream); // -> [4,2,224,224]
    #undef P
}
